// SelfAttn_8211977470143
// MI455X (gfx1250) — matrix operand order repaired and checked by reading, not compiled, not run on hardware
//
#include <hip/hip_runtime.h>
#include <hip/hip_fp16.h>

typedef __attribute__((ext_vector_type(16))) _Float16 v16h;
typedef __attribute__((ext_vector_type(8)))  float    v8f;

#define DIM    512
#define NHEAD  8
#define HDIM   64
#define SEQ    4096
#define BATCH  2
#define MTOT   (BATCH * SEQ)     // 8192
#define SCALE  0.125f            // 1/sqrt(64)

// ---------------------------------------------------------------------------
// CDNA5 async copy: 16 bytes per lane, global -> LDS, tracked by ASYNCcnt.
// VDST operand is the wave-relative LDS byte offset (low 32 bits of the flat
// shared-aperture address).
// ---------------------------------------------------------------------------
__device__ __forceinline__ void async_load_b128(const void* gptr, void* lptr) {
    unsigned int lds = (unsigned int)(unsigned long long)lptr;
    unsigned long long ga = (unsigned long long)gptr;
    asm volatile("global_load_async_to_lds_b128 %0, %1, off"
                 :: "v"(lds), "v"(ga)
                 : "memory");
}
__device__ __forceinline__ void wait_async() {
    asm volatile("s_wait_asynccnt 0x0" ::: "memory");
}

// ---------------------------------------------------------------------------
// WMMA fragment loaders (wave32, A and B both use contiguous k = 16*(lane/16) + s)
// ---------------------------------------------------------------------------

// A-matrix 16x32 f16 fragment from row-major LDS tile [rows][stride].
__device__ __forceinline__ v16h load_a_frag(const _Float16* lds, int row0, int k0, int stride) {
    const int lane = threadIdx.x & 31;
    const int half = lane >> 4;
    const _Float16* p = lds + (size_t)(row0 + (lane & 15)) * stride + k0;
    v16h a;
#pragma unroll
    for (int i = 0; i < 4; ++i) {
        int klo = half * 16 + 2 * i;
        a[2 * i]     = p[klo];
        a[2 * i + 1] = p[klo + 1];
        int khi = half * 16 + 8 + 2 * i;
        a[8 + 2 * i]     = p[khi];
        a[8 + 2 * i + 1] = p[khi + 1];
    }
    return a;
}

// B-matrix 32x16 f16 fragment, LDS tile stored n-major: [n][stride] over k.
__device__ __forceinline__ v16h load_b_frag_nmajor(const _Float16* lds, int n0, int k0, int stride) {
    const int lane = threadIdx.x & 31;
    const int half = lane >> 4;
    const _Float16* p = lds + (size_t)(n0 + (lane & 15)) * stride + k0 + half * 16;
    v16h b;
#pragma unroll
    for (int i = 0; i < 8; ++i) {
        b[2 * i]     = p[2 * i];
        b[2 * i + 1] = p[2 * i + 1];
    }
    return b;
}

// B-matrix 32x16 f16 fragment, LDS tile stored k-major: [k][stride] over n.
__device__ __forceinline__ v16h load_b_frag_kmajor(const _Float16* lds, int k0, int n0, int stride) {
    const int lane = threadIdx.x & 31;
    const int half = lane >> 4;
    const _Float16* p = lds + (size_t)(k0 + half * 16) * stride + n0 + (lane & 15);
    v16h b;
#pragma unroll
    for (int i = 0; i < 8; ++i) {
        b[2 * i]     = p[(2 * i) * stride];
        b[2 * i + 1] = p[(2 * i + 1) * stride];
    }
    return b;
}

__device__ __forceinline__ v8f wmma_f16(v16h a, v16h b, v8f c) {
    return __builtin_amdgcn_wmma_f32_16x16x32_f16(false, a, false, b, (short)0, c, false, false);
}

// ---------------------------------------------------------------------------
// fp32 -> f16 conversion (all sizes are multiples of 1024)
// ---------------------------------------------------------------------------
__global__ void cvt_f32_f16(const float* __restrict__ in, _Float16* __restrict__ out, int n) {
    int i = (blockIdx.x * blockDim.x + threadIdx.x) * 4;
    if (i + 3 < n) {
        float4 v = *(const float4*)(in + i);
        out[i]     = (_Float16)v.x;
        out[i + 1] = (_Float16)v.y;
        out[i + 2] = (_Float16)v.z;
        out[i + 3] = (_Float16)v.w;
    }
}

// ---------------------------------------------------------------------------
// QKV projection: [8192,512] x W[1536,512]^T -> Q/K/V [b,h,n,d] f16
// Block: 256 threads (8 waves), block tile 64(M) x 256(N), wave tile 32x64.
// A/B tiles staged via global_load_async_to_lds_b128.
// ---------------------------------------------------------------------------
__global__ void __launch_bounds__(256)
qkv_gemm(const _Float16* __restrict__ X, const _Float16* __restrict__ W,
         _Float16* __restrict__ Q, _Float16* __restrict__ Kd, _Float16* __restrict__ V) {
    __shared__ _Float16 As[64][40];    // [m][k]   (row = 80B, 16B aligned)
    __shared__ _Float16 Bs[256][40];   // [f][k]   (n-major)
    const int m0   = blockIdx.x * 64;
    const int f0   = blockIdx.y * 256;
    const int tid  = threadIdx.x;
    const int wave = tid >> 5;
    const int wm   = (wave >> 2) * 32;   // 0 or 32
    const int wn   = (wave & 3) * 64;    // 0..192

    const int ar = tid >> 2;             // A tile row this thread stages
    const int ac = (tid & 3) * 8;        // A tile col (f16)
    const int br = tid;                  // B tile row (= output feature)

    v8f acc[2][4] = {};

    for (int k0 = 0; k0 < DIM; k0 += 32) {
        __syncthreads();
        // A tile: 64x32, one b128 per thread
        async_load_b128(X + (size_t)(m0 + ar) * DIM + k0 + ac, &As[ar][ac]);
        // B tile: 256x32, four b128 per thread (one row each)
#pragma unroll
        for (int j = 0; j < 4; ++j)
            async_load_b128(W + (size_t)(f0 + br) * DIM + k0 + j * 8, &Bs[br][j * 8]);
        if (k0 + 32 < DIM) {
            __builtin_prefetch(X + (size_t)(m0 + ar) * DIM + k0 + 32, 0, 1);
            __builtin_prefetch(W + (size_t)(f0 + br) * DIM + k0 + 32, 0, 1);
        }
        wait_async();
        __syncthreads();

        v16h a0 = load_a_frag(&As[0][0], wm,      0, 40);
        v16h a1 = load_a_frag(&As[0][0], wm + 16, 0, 40);
#pragma unroll
        for (int ni = 0; ni < 4; ++ni) {
            v16h b = load_b_frag_nmajor(&Bs[0][0], wn + ni * 16, 0, 40);
            acc[0][ni] = wmma_f16(a0, b, acc[0][ni]);
            acc[1][ni] = wmma_f16(a1, b, acc[1][ni]);
        }
    }

    // Scatter into per-head Q/K/V [b,h,n,d]
    const int lane  = tid & 31;
    const int col   = lane & 15;
    const int rbase = (lane >> 4) * 8;
#pragma unroll
    for (int ni = 0; ni < 4; ++ni) {
        int f = f0 + wn + ni * 16 + col;
        _Float16* base = (f < DIM) ? Q : (f < 2 * DIM) ? Kd : V;
        int fh = f & (DIM - 1);
        int h  = fh >> 6;
        int d  = fh & (HDIM - 1);
#pragma unroll
        for (int mi = 0; mi < 2; ++mi) {
#pragma unroll
            for (int i = 0; i < 8; ++i) {
                int m  = m0 + wm + mi * 16 + rbase + i;
                int bb = m >> 12;           // / SEQ
                int nn = m & (SEQ - 1);
                base[((size_t)(bb * NHEAD + h) * SEQ + nn) * HDIM + d] = (_Float16)acc[mi][ni][i];
            }
        }
    }
}

// ---------------------------------------------------------------------------
// Flash attention: grid (SEQ/64, BATCH*NHEAD), 128 threads = 4 waves.
// Each wave owns 16 query rows; K/V streamed in 64-key tiles through LDS
// via async-to-LDS copies; online softmax; P restaged via LDS for P*V.
// ---------------------------------------------------------------------------
__global__ void __launch_bounds__(128)
flash_attn(const _Float16* __restrict__ Q, const _Float16* __restrict__ K,
           const _Float16* __restrict__ V, _Float16* __restrict__ O) {
    __shared__ _Float16 Qs[64][72];
    __shared__ _Float16 Ks[64][72];
    __shared__ _Float16 Vs[64][72];
    __shared__ _Float16 Ps[4][16][72];   // per-wave probability tile

    const int q0  = blockIdx.x * 64;
    const int bh  = blockIdx.y;
    const int b   = bh >> 3;
    const int h   = bh & 7;
    const _Float16* Qh = Q + (size_t)bh * SEQ * HDIM;
    const _Float16* Kh = K + (size_t)bh * SEQ * HDIM;
    const _Float16* Vh = V + (size_t)bh * SEQ * HDIM;

    const int tid   = threadIdx.x;
    const int wave  = tid >> 5;
    const int lane  = tid & 31;
    const int col   = lane & 15;
    const int rbase = (lane >> 4) * 8;

    // Load Q tile (64x64): 128 threads x 4 async b128 chunks
#pragma unroll
    for (int j = 0; j < 4; ++j) {
        int idx = tid + j * 128;
        int r = idx >> 3;
        int c = (idx & 7) * 8;
        async_load_b128(Qh + (size_t)(q0 + r) * HDIM + c, &Qs[r][c]);
    }
    wait_async();
    __syncthreads();

    v16h qa0 = load_a_frag(&Qs[0][0], wave * 16, 0, 72);
    v16h qa1 = load_a_frag(&Qs[0][0], wave * 16, 32, 72);

    float m_i[8], l_i[8];
    v8f o_acc[4] = {};
#pragma unroll
    for (int i = 0; i < 8; ++i) { m_i[i] = -1e30f; l_i[i] = 0.0f; }

    for (int kt = 0; kt < SEQ / 64; ++kt) {
        __syncthreads();   // previous iter's LDS reads done
#pragma unroll
        for (int j = 0; j < 4; ++j) {
            int idx = tid + j * 128;
            int r = idx >> 3;
            int c = (idx & 7) * 8;
            async_load_b128(Kh + (size_t)(kt * 64 + r) * HDIM + c, &Ks[r][c]);
            async_load_b128(Vh + (size_t)(kt * 64 + r) * HDIM + c, &Vs[r][c]);
        }
        wait_async();
        __syncthreads();

        // S = Q K^T (16 q-rows x 64 keys), 4 n-tiles
        v8f s[4];
#pragma unroll
        for (int nt = 0; nt < 4; ++nt) {
            v16h kb0 = load_b_frag_nmajor(&Ks[0][0], nt * 16, 0, 72);
            v16h kb1 = load_b_frag_nmajor(&Ks[0][0], nt * 16, 32, 72);
            v8f z = {};
            z = wmma_f16(qa0, kb0, z);
            z = wmma_f16(qa1, kb1, z);
            s[nt] = z;
        }

        // Online softmax: row = rbase + i per lane-half; reduce within 16-lane group
        float mnew[8], alpha[8];
#pragma unroll
        for (int i = 0; i < 8; ++i) {
            float cand = -1e30f;
#pragma unroll
            for (int nt = 0; nt < 4; ++nt) cand = fmaxf(cand, s[nt][i] * SCALE);
#pragma unroll
            for (int off = 1; off < 16; off <<= 1)
                cand = fmaxf(cand, __shfl_xor(cand, off, 32));
            mnew[i]  = fmaxf(m_i[i], cand);
            alpha[i] = __expf(m_i[i] - mnew[i]);
            m_i[i]   = mnew[i];
        }
#pragma unroll
        for (int i = 0; i < 8; ++i) {
            float rs = 0.0f;
#pragma unroll
            for (int nt = 0; nt < 4; ++nt) {
                float p = __expf(s[nt][i] * SCALE - mnew[i]);
                s[nt][i] = p;
                rs += p;
            }
#pragma unroll
            for (int off = 1; off < 16; off <<= 1)
                rs += __shfl_xor(rs, off, 32);
            l_i[i] = l_i[i] * alpha[i] + rs;
#pragma unroll
            for (int nt = 0; nt < 4; ++nt)
                o_acc[nt][i] = o_acc[nt][i] * alpha[i];
        }

        // Stage P (f16) in per-wave LDS, re-read as A fragments
#pragma unroll
        for (int nt = 0; nt < 4; ++nt)
#pragma unroll
            for (int i = 0; i < 8; ++i)
                Ps[wave][rbase + i][nt * 16 + col] = (_Float16)s[nt][i];

        v16h pa0 = load_a_frag(&Ps[wave][0][0], 0, 0, 72);
        v16h pa1 = load_a_frag(&Ps[wave][0][0], 0, 32, 72);
#pragma unroll
        for (int nt = 0; nt < 4; ++nt) {
            v16h vb0 = load_b_frag_kmajor(&Vs[0][0], 0,  nt * 16, 72);
            v16h vb1 = load_b_frag_kmajor(&Vs[0][0], 32, nt * 16, 72);
            o_acc[nt] = wmma_f16(pa0, vb0, o_acc[nt]);
            o_acc[nt] = wmma_f16(pa1, vb1, o_acc[nt]);
        }
    }

    // Normalize and write O as f16 [b, n, h*64+d]
    float rinv[8];
#pragma unroll
    for (int i = 0; i < 8; ++i) rinv[i] = 1.0f / l_i[i];
#pragma unroll
    for (int nt = 0; nt < 4; ++nt) {
        int d = nt * 16 + col;
#pragma unroll
        for (int i = 0; i < 8; ++i) {
            int qrow = q0 + wave * 16 + rbase + i;
            O[((size_t)b * SEQ + qrow) * DIM + h * HDIM + d] = (_Float16)(o_acc[nt][i] * rinv[i]);
        }
    }
}

// ---------------------------------------------------------------------------
// Output projection: O[8192,512] x Wout[512,512]^T + bias -> fp32 out
// Block tile 64x256, wave tile 32x64, async LDS staging.
// ---------------------------------------------------------------------------
__global__ void __launch_bounds__(256)
out_gemm(const _Float16* __restrict__ A, const _Float16* __restrict__ W,
         const float* __restrict__ bias, float* __restrict__ out) {
    __shared__ _Float16 As[64][40];
    __shared__ _Float16 Bs[256][40];
    const int m0   = blockIdx.x * 64;
    const int f0   = blockIdx.y * 256;
    const int tid  = threadIdx.x;
    const int wave = tid >> 5;
    const int wm   = (wave >> 2) * 32;
    const int wn   = (wave & 3) * 64;

    const int ar = tid >> 2;
    const int ac = (tid & 3) * 8;
    const int br = tid;

    v8f acc[2][4] = {};

    for (int k0 = 0; k0 < DIM; k0 += 32) {
        __syncthreads();
        async_load_b128(A + (size_t)(m0 + ar) * DIM + k0 + ac, &As[ar][ac]);
#pragma unroll
        for (int j = 0; j < 4; ++j)
            async_load_b128(W + (size_t)(f0 + br) * DIM + k0 + j * 8, &Bs[br][j * 8]);
        wait_async();
        __syncthreads();

        v16h a0 = load_a_frag(&As[0][0], wm,      0, 40);
        v16h a1 = load_a_frag(&As[0][0], wm + 16, 0, 40);
#pragma unroll
        for (int ni = 0; ni < 4; ++ni) {
            v16h b = load_b_frag_nmajor(&Bs[0][0], wn + ni * 16, 0, 40);
            acc[0][ni] = wmma_f16(a0, b, acc[0][ni]);
            acc[1][ni] = wmma_f16(a1, b, acc[1][ni]);
        }
    }

    const int lane  = tid & 31;
    const int col   = lane & 15;
    const int rbase = (lane >> 4) * 8;
#pragma unroll
    for (int ni = 0; ni < 4; ++ni) {
        int f = f0 + wn + ni * 16 + col;
        float bv = bias[f];
#pragma unroll
        for (int mi = 0; mi < 2; ++mi) {
#pragma unroll
            for (int i = 0; i < 8; ++i) {
                int m = m0 + wm + mi * 16 + rbase + i;
                out[(size_t)m * DIM + f] = acc[mi][ni][i] + bv;
            }
        }
    }
}

// ---------------------------------------------------------------------------
// Host-side launch
// ---------------------------------------------------------------------------
extern "C" void kernel_launch(void* const* d_in, const int* in_sizes, int n_in,
                              void* d_out, int out_size, void* d_ws, size_t ws_size,
                              hipStream_t stream) {
    const float* x     = (const float*)d_in[0];
    const float* w_qkv = (const float*)d_in[1];
    const float* w_out = (const float*)d_in[2];
    const float* b_out = (const float*)d_in[3];
    float* out = (float*)d_out;

    char* ws = (char*)d_ws;
    // f16 workspace layout (bytes)
    _Float16* Xh    = (_Float16*)(ws + 0);          // 8192*512   = 8,388,608 B
    _Float16* Wqkvh = (_Float16*)(ws + 8388608);    // 1536*512   = 1,572,864 B
    _Float16* Wouth = (_Float16*)(ws + 9961472);    // 512*512    =   524,288 B
    _Float16* Qh    = (_Float16*)(ws + 10485760);   // 2*8*4096*64 = 8,388,608 B
    _Float16* Kh    = (_Float16*)(ws + 18874368);
    _Float16* Vh    = (_Float16*)(ws + 27262976);
    _Float16* Oh    = (_Float16*)(ws + 35651584);   // end ~44 MB

    cvt_f32_f16<<<MTOT * DIM / 1024, 256, 0, stream>>>(x, Xh, MTOT * DIM);
    cvt_f32_f16<<<3 * DIM * DIM / 1024, 256, 0, stream>>>(w_qkv, Wqkvh, 3 * DIM * DIM);
    cvt_f32_f16<<<DIM * DIM / 1024, 256, 0, stream>>>(w_out, Wouth, DIM * DIM);

    qkv_gemm<<<dim3(MTOT / 64, 3 * DIM / 256), 256, 0, stream>>>(Xh, Wqkvh, Qh, Kh, Vh);
    flash_attn<<<dim3(SEQ / 64, BATCH * NHEAD), 128, 0, stream>>>(Qh, Kh, Vh, Oh);
    out_gemm<<<dim3(MTOT / 64, DIM / 256), 256, 0, stream>>>(Oh, Wouth, b_out, out);
}
